// Deep_OSTTP_Model_25829933318903
// MI455X (gfx1250) — compile-verified, matrix operand order checked by reading, not run on hardware
//
#include <hip/hip_runtime.h>
#include <hip/hip_bf16.h>
#include <math.h>

#define BATCH   2048
#define HDIM    2048
#define ODIM    1024
#define NLAYERS 4
#define NSTEPS  8

typedef __attribute__((ext_vector_type(16))) __bf16       v16bf;
typedef __attribute__((ext_vector_type(8)))  float        v8f;
typedef __attribute__((ext_vector_type(8)))  __bf16       bf16x8;
typedef __attribute__((ext_vector_type(4)))  float        f32x4;
typedef __attribute__((ext_vector_type(4)))  unsigned int u32x4;
typedef int i32x4 __attribute__((vector_size(16)));   // matches builtin param type

union Frag { v16bf v; u32x4 q[2]; };

constexpr int LDT = 40;   // padded LDS row stride (elements): conflict-free b128 frag loads

#if defined(__gfx1250__) && __has_builtin(__builtin_amdgcn_global_load_async_to_lds_b128)
#define USE_ASYNC_LDS 1
#else
#define USE_ASYNC_LDS 0
#endif

#if USE_ASYNC_LDS
__device__ __forceinline__ void async_cp16(const __bf16* g, __bf16* l) {
  __builtin_amdgcn_global_load_async_to_lds_b128(
      (__attribute__((address_space(1))) i32x4*)g,
      (__attribute__((address_space(3))) i32x4*)l, 0, 0);
}
__device__ __forceinline__ void async_wait0() {
#if __has_builtin(__builtin_amdgcn_s_wait_asynccnt)
  __builtin_amdgcn_s_wait_asynccnt(0);
#else
  asm volatile("s_wait_asynccnt 0x0" ::: "memory");
#endif
}
#endif

// ---------------------------------------------------------------------------
// f32 -> bf16 conversion (8 elements / thread)
// ---------------------------------------------------------------------------
__global__ __launch_bounds__(256) void cvt_bf16_kernel(
    const float* __restrict__ in, __bf16* __restrict__ out, int n8)
{
  int i = blockIdx.x * blockDim.x + threadIdx.x;
  if (i >= n8) return;
  const f32x4* p = (const f32x4*)in + (size_t)i * 2;
  f32x4 f0 = p[0], f1 = p[1];
  bf16x8 o;
  o[0] = (__bf16)f0[0]; o[1] = (__bf16)f0[1]; o[2] = (__bf16)f0[2]; o[3] = (__bf16)f0[3];
  o[4] = (__bf16)f1[0]; o[5] = (__bf16)f1[1]; o[6] = (__bf16)f1[2]; o[7] = (__bf16)f1[3];
  *((bf16x8*)out + (size_t)i) = o;
}

// ---------------------------------------------------------------------------
// first recurrent step: h0 == 0  =>  h1 = tanh(pre + bz)   (exact algebra)
// ---------------------------------------------------------------------------
__global__ __launch_bounds__(256) void first_step_kernel(
    const float* __restrict__ pre, const float* __restrict__ bz,
    __bf16* __restrict__ h)
{
  size_t i = (size_t)blockIdx.x * blockDim.x + threadIdx.x;  // over BATCH*HDIM
  int col = (int)(i & (size_t)(HDIM - 1));
  h[i] = (__bf16)tanhf(pre[i] + bz[col]);
}

// ---------------------------------------------------------------------------
// C[M,N] = epilogue( A[M,K](bf16) * B[N,K](bf16)^T )
// 128x128 block tile, 8 waves, wave tile 32x64 (2x4 wmma 16x16x32 bf16 tiles)
// Tile staging uses GLOBAL_LOAD_ASYNC_TO_LDS_B128 (ASYNCcnt) when available.
// ---------------------------------------------------------------------------
template <bool ADD_BIAS, bool ADD_PRE, bool DO_TANH, bool ST_F32, bool ST_BF16>
__global__ __launch_bounds__(256, 1) void gemm_bf16_kernel(
    const __bf16* __restrict__ A,    // [M,K] row-major
    const __bf16* __restrict__ Bw,   // [N,K] row-major (i.e. op is A * Bw^T)
    const float*  __restrict__ bias, // [N]
    const float*  __restrict__ pre,  // [M,N]
    float*   __restrict__ outF,      // [M,N]
    __bf16*  __restrict__ outB,      // [M,N]
    int M, int N, int K)
{
  __shared__ __bf16 sA[2][128 * LDT];
  __shared__ __bf16 sB[2][128 * LDT];

  const int t    = threadIdx.x;
  const int lane = t & 31;
  const int wv   = t >> 5;
  const int wM   = wv & 3;    // 4 wave rows * 32 = 128 M
  const int wN   = wv >> 2;   // 2 wave cols * 64 = 128 N
  const int lr   = lane & 15; // lane row/col within 16
  const int hi   = lane >> 4; // half-wave select

  const int mBlk = blockIdx.y * 128;
  const int nBlk = blockIdx.x * 128;

  const __bf16* Ab = A  + (size_t)mBlk * K;
  const __bf16* Bb = Bw + (size_t)nBlk * K;

  // tile staging: 128 rows x 32 cols bf16 = 512 x 16B chunks, 2 per thread
  const int c0  = t,        c1  = t + 256;
  const int ar0 = c0 >> 2,  ac0 = (c0 & 3) * 8;
  const int ar1 = c1 >> 2,  ac1 = (c1 & 3) * 8;

  const int KT = K >> 5;   // K-steps of 32

  v8f acc[2][4] = {};

  // prologue: stage k-tile 0
  {
#if USE_ASYNC_LDS
    async_cp16(Ab + (size_t)ar0 * K + ac0, &sA[0][ar0 * LDT + ac0]);
    async_cp16(Ab + (size_t)ar1 * K + ac1, &sA[0][ar1 * LDT + ac1]);
    async_cp16(Bb + (size_t)ar0 * K + ac0, &sB[0][ar0 * LDT + ac0]);
    async_cp16(Bb + (size_t)ar1 * K + ac1, &sB[0][ar1 * LDT + ac1]);
    async_wait0();
#else
    u32x4 a0 = *(const u32x4*)(Ab + (size_t)ar0 * K + ac0);
    u32x4 a1 = *(const u32x4*)(Ab + (size_t)ar1 * K + ac1);
    u32x4 b0 = *(const u32x4*)(Bb + (size_t)ar0 * K + ac0);
    u32x4 b1 = *(const u32x4*)(Bb + (size_t)ar1 * K + ac1);
    *(u32x4*)&sA[0][ar0 * LDT + ac0] = a0;
    *(u32x4*)&sA[0][ar1 * LDT + ac1] = a1;
    *(u32x4*)&sB[0][ar0 * LDT + ac0] = b0;
    *(u32x4*)&sB[0][ar1 * LDT + ac1] = b1;
#endif
  }
  __syncthreads();

  for (int kt = 0; kt < KT; ++kt) {
    const int  buf  = kt & 1;
    const bool more = (kt + 1) < KT;

#if USE_ASYNC_LDS
    if (more) {  // DMA next k-tile cache->LDS while this one computes
      const size_t ko = (size_t)(kt + 1) * 32;
      async_cp16(Ab + (size_t)ar0 * K + ko + ac0, &sA[buf ^ 1][ar0 * LDT + ac0]);
      async_cp16(Ab + (size_t)ar1 * K + ko + ac1, &sA[buf ^ 1][ar1 * LDT + ac1]);
      async_cp16(Bb + (size_t)ar0 * K + ko + ac0, &sB[buf ^ 1][ar0 * LDT + ac0]);
      async_cp16(Bb + (size_t)ar1 * K + ko + ac1, &sB[buf ^ 1][ar1 * LDT + ac1]);
    }
#else
    u32x4 a0, a1, b0, b1;
    if (more) {  // prefetch next k-tile into VGPRs while this one computes
      const size_t ko = (size_t)(kt + 1) * 32;
      a0 = *(const u32x4*)(Ab + (size_t)ar0 * K + ko + ac0);
      a1 = *(const u32x4*)(Ab + (size_t)ar1 * K + ko + ac1);
      b0 = *(const u32x4*)(Bb + (size_t)ar0 * K + ko + ac0);
      b1 = *(const u32x4*)(Bb + (size_t)ar1 * K + ko + ac1);
    }
#endif

    // fragment loads per documented 16-bit A/B lane layout:
    // lane L (L<16): row M=L, K chunks {0..7, 16..23}; (L>=16): M=L-16, K {8..15, 24..31}
    Frag fa[2], fb[4];
#pragma unroll
    for (int i = 0; i < 2; ++i) {
      const __bf16* p = &sA[buf][(wM * 32 + i * 16 + lr) * LDT + hi * 8];
      fa[i].q[0] = *(const u32x4*)p;
      fa[i].q[1] = *(const u32x4*)(p + 16);
    }
#pragma unroll
    for (int j = 0; j < 4; ++j) {
      const __bf16* p = &sB[buf][(wN * 64 + j * 16 + lr) * LDT + hi * 8];
      fb[j].q[0] = *(const u32x4*)p;
      fb[j].q[1] = *(const u32x4*)(p + 16);
    }

#pragma unroll
    for (int i = 0; i < 2; ++i)
#pragma unroll
      for (int j = 0; j < 4; ++j)
        acc[i][j] = __builtin_amdgcn_wmma_f32_16x16x32_bf16(
            false, fa[i].v, false, fb[j].v, (short)0, acc[i][j], false, false);

#if USE_ASYNC_LDS
    if (more) async_wait0();
#else
    if (more) {
      *(u32x4*)&sA[buf ^ 1][ar0 * LDT + ac0] = a0;
      *(u32x4*)&sA[buf ^ 1][ar1 * LDT + ac1] = a1;
      *(u32x4*)&sB[buf ^ 1][ar0 * LDT + ac0] = b0;
      *(u32x4*)&sB[buf ^ 1][ar1 * LDT + ac1] = b1;
    }
#endif
    __syncthreads();
  }

  // epilogue: C/D layout: VGPR r, lane L -> (m = r + 8*(L/16), n = L%16)
#pragma unroll
  for (int j = 0; j < 4; ++j) {
    const int n = nBlk + wN * 64 + j * 16 + lr;
    const float bv = ADD_BIAS ? bias[n] : 0.0f;
#pragma unroll
    for (int i = 0; i < 2; ++i) {
      const int mb = mBlk + wM * 32 + i * 16 + hi * 8;
#pragma unroll
      for (int r = 0; r < 8; ++r) {
        const int m = mb + r;
        float v = acc[i][j][r];
        if (ADD_BIAS) v += bv;
        if (ADD_PRE)  v += pre[(size_t)m * N + n];
        if (DO_TANH)  v = tanhf(v);
        if (ST_F32)   outF[(size_t)m * N + n] = v;
        if (ST_BF16)  outB[(size_t)m * N + n] = (__bf16)v;
      }
    }
  }
}

// ---------------------------------------------------------------------------
// launch
// ---------------------------------------------------------------------------
extern "C" void kernel_launch(void* const* d_in, const int* in_sizes, int n_in,
                              void* d_out, int out_size, void* d_ws, size_t ws_size,
                              hipStream_t stream)
{
  const float* x      = (const float*)d_in[0];
  const float* W_in   = (const float*)d_in[1];
  const float* b_in   = (const float*)d_in[2];
  const float* Wz     = (const float*)d_in[3];
  const float* bz     = (const float*)d_in[4];
  const float* Wx     = (const float*)d_in[5];
  /* d_in[6] = R, unused in forward */
  const float* W_head = (const float*)d_in[7];
  const float* b_head = (const float*)d_in[8];
  float* out = (float*)d_out;

  // workspace suballocation (256B aligned); total ~116 MB -> resident in L2
  char* w = (char*)d_ws;
  auto suballoc = [&](size_t bytes) -> char* {
    char* p = w; w += (bytes + 255) & ~(size_t)255; return p;
  };
  __bf16* xb   = (__bf16*)suballoc((size_t)BATCH * HDIM * 2);
  __bf16* Winb = (__bf16*)suballoc((size_t)HDIM * HDIM * 2);
  __bf16* Wzb  = (__bf16*)suballoc((size_t)NLAYERS * HDIM * HDIM * 2);
  __bf16* Wxb  = (__bf16*)suballoc((size_t)NLAYERS * HDIM * HDIM * 2);
  __bf16* Whb  = (__bf16*)suballoc((size_t)ODIM * HDIM * 2);
  __bf16* hA   = (__bf16*)suballoc((size_t)BATCH * HDIM * 2);
  __bf16* hB   = (__bf16*)suballoc((size_t)BATCH * HDIM * 2);
  float*  preF = (float*)suballoc((size_t)BATCH * HDIM * 4);

  auto cvt = [&](const float* src, __bf16* dst, size_t n) {
    int n8 = (int)(n / 8);
    cvt_bf16_kernel<<<(n8 + 255) / 256, 256, 0, stream>>>(src, dst, n8);
  };
  cvt(x,      xb,   (size_t)BATCH * HDIM);
  cvt(W_in,   Winb, (size_t)HDIM * HDIM);
  cvt(Wz,     Wzb,  (size_t)NLAYERS * HDIM * HDIM);
  cvt(Wx,     Wxb,  (size_t)NLAYERS * HDIM * HDIM);
  cvt(W_head, Whb,  (size_t)ODIM * HDIM);

  const dim3 blk(256);
  const dim3 grdHH(HDIM / 128, BATCH / 128);

  // z = x @ W_in^T + b_in   -> bf16
  gemm_bf16_kernel<true, false, false, false, true><<<grdHH, blk, 0, stream>>>(
      xb, Winb, b_in, nullptr, nullptr, hA, BATCH, HDIM, HDIM);

  __bf16* cur = hA;
  for (int l = 0; l < NLAYERS; ++l) {
    // pre = z @ Wx[l]^T  -> f32
    gemm_bf16_kernel<false, false, false, true, false><<<grdHH, blk, 0, stream>>>(
        cur, Wxb + (size_t)l * HDIM * HDIM, nullptr, nullptr, preF, nullptr,
        BATCH, HDIM, HDIM);
    // step 1 (h0 == 0): h1 = tanh(pre + bz)
    __bf16* h = (cur == hA) ? hB : hA;
    first_step_kernel<<<(BATCH * HDIM) / 256, 256, 0, stream>>>(
        preF, bz + (size_t)l * HDIM, h);
    // steps 2..8: h = tanh(h @ Wz[l]^T + bz + pre)
    for (int s = 1; s < NSTEPS; ++s) {
      __bf16* nxt = (h == hA) ? hB : hA;
      gemm_bf16_kernel<true, true, true, false, true><<<grdHH, blk, 0, stream>>>(
          h, Wzb + (size_t)l * HDIM * HDIM, bz + (size_t)l * HDIM, preF,
          nullptr, nxt, BATCH, HDIM, HDIM);
      h = nxt;
    }
    cur = h;
  }

  // out = z @ W_head^T + b_head  -> f32
  const dim3 grdHead(ODIM / 128, BATCH / 128);
  gemm_bf16_kernel<true, false, false, true, false><<<grdHead, blk, 0, stream>>>(
      cur, Whb, b_head, nullptr, out, nullptr, BATCH, ODIM, HDIM);
}
